// Module_63513976373722
// MI455X (gfx1250) — matrix operand order repaired and checked by reading, not compiled, not run on hardware
//
#include <hip/hip_runtime.h>
#include <math.h>

// ---------------------------------------------------------------------------
// Problem constants (match reference)
// ---------------------------------------------------------------------------
#define B_   32
#define L_   2048
#define D_   1024
#define H_   16
#define HD_  64

// Main-kernel blocking
#define RB          128            // L-rows per workgroup (within one b)
#define KC          32             // d (reduction) chunk per WMMA stage
#define NKC         (D_ / KC)      // 32 chunks
#define JT_TOTAL    (D_ / 16)      // 64 j-tiles of 16
#define JBLK        512            // j-columns per workgroup (8 heads)
#define NJB         2
#define JT_PER_BLK  32
#define MAIN_THREADS 1024          // 32 waves (wave32)

typedef __attribute__((ext_vector_type(16))) __bf16 v16bf;
typedef __attribute__((ext_vector_type(2)))  __bf16 v2bf;
typedef __attribute__((ext_vector_type(8)))  float  v8f;
typedef __attribute__((ext_vector_type(4)))  unsigned int u32x4;
typedef __attribute__((ext_vector_type(8)))  int i32x8;
typedef __attribute__((ext_vector_type(4)))  int i32x4;

#if defined(__HIP_DEVICE_COMPILE__) && __has_builtin(__builtin_amdgcn_tensor_load_to_lds) && \
    __has_builtin(__builtin_amdgcn_s_wait_tensorcnt)
#define USE_TDM 1
#else
#define USE_TDM 0
#endif

union BF16Frag {
    v16bf v;
    v2bf h[8];
    unsigned int w[8];
    unsigned short u[16];
    uint4 q[2];
};

static __device__ __forceinline__ unsigned short f32_to_bf16(float f) {
    union { float f; unsigned int u; } x;
    x.f = f;
    unsigned int u = x.u;
    u += 0x7FFFu + ((u >> 16) & 1u);     // round-to-nearest-even
    return (unsigned short)(u >> 16);
}

// Pack two f32 -> two bf16 (round-half-up) in one v_perm_b32 + two v_add:
// result[15:0] = bf16(a), result[31:16] = bf16(b)
static __device__ __forceinline__ unsigned int pack_bf16_rn(float a, float b) {
    unsigned int ua = __float_as_uint(a) + 0x8000u;
    unsigned int ub = __float_as_uint(b) + 0x8000u;
    return __byte_perm(ua, ub, 0x07060302);   // {ub[3],ub[2],ua[3],ua[2]}
}

// Build a 16x32 bf16 A-fragment from an LDS row of 32 f32.
// ISA 16-bit A 16x32 layout: lane (hi half selects K offset):
//   e in [0,8):  K = e + 8*hi          (contiguous run at 8*hi)
//   e in [8,16): K = (e-8) + 16 + 8*hi (contiguous run at 16+8*hi)
static __device__ __forceinline__ v16bf build_a_frag(const float* row, int hi) {
    BF16Frag A;
    const float4 f0 = *(const float4*)(row + 8 * hi);
    const float4 f1 = *(const float4*)(row + 8 * hi + 4);
    const float4 f2 = *(const float4*)(row + 16 + 8 * hi);
    const float4 f3 = *(const float4*)(row + 16 + 8 * hi + 4);
#if defined(__HIP_DEVICE_COMPILE__) && __has_builtin(__builtin_amdgcn_cvt_pk_bf16_f32)
    A.h[0] = __builtin_amdgcn_cvt_pk_bf16_f32(f0.x, f0.y);
    A.h[1] = __builtin_amdgcn_cvt_pk_bf16_f32(f0.z, f0.w);
    A.h[2] = __builtin_amdgcn_cvt_pk_bf16_f32(f1.x, f1.y);
    A.h[3] = __builtin_amdgcn_cvt_pk_bf16_f32(f1.z, f1.w);
    A.h[4] = __builtin_amdgcn_cvt_pk_bf16_f32(f2.x, f2.y);
    A.h[5] = __builtin_amdgcn_cvt_pk_bf16_f32(f2.z, f2.w);
    A.h[6] = __builtin_amdgcn_cvt_pk_bf16_f32(f3.x, f3.y);
    A.h[7] = __builtin_amdgcn_cvt_pk_bf16_f32(f3.z, f3.w);
#else
    A.w[0] = pack_bf16_rn(f0.x, f0.y);
    A.w[1] = pack_bf16_rn(f0.z, f0.w);
    A.w[2] = pack_bf16_rn(f1.x, f1.y);
    A.w[3] = pack_bf16_rn(f1.z, f1.w);
    A.w[4] = pack_bf16_rn(f2.x, f2.y);
    A.w[5] = pack_bf16_rn(f2.z, f2.w);
    A.w[6] = pack_bf16_rn(f3.x, f3.y);
    A.w[7] = pack_bf16_rn(f3.z, f3.w);
#endif
    return A.v;
}

// One k-chunk GEMM step for a wave: 8 B-fragments (immediate offsets off bptr),
// loaded in two groups of 4 around the A build for load/WMMA overlap.
static __device__ __forceinline__ void gemm_step(const unsigned short* bptr,
                                                 const float* slabrow, int hi,
                                                 v8f* acc) {
    uint4 ba[8], bb[8];
#pragma unroll
    for (int i = 0; i < 4; ++i) {
        const uint4* bp = (const uint4*)(bptr + i * 512);
        ba[2 * i] = bp[0];
        ba[2 * i + 1] = bp[1];
    }
    const v16bf A = build_a_frag(slabrow, hi);   // ds loads + VALU overlap vmem
#pragma unroll
    for (int i = 0; i < 4; ++i) {
        const uint4* bp = (const uint4*)(bptr + (i + 4) * 512);
        bb[2 * i] = bp[0];
        bb[2 * i + 1] = bp[1];
    }
#pragma unroll
    for (int i = 0; i < 4; ++i) {
        BF16Frag Bf;
        Bf.q[0] = ba[2 * i];
        Bf.q[1] = ba[2 * i + 1];
        acc[i] = __builtin_amdgcn_wmma_f32_16x16x32_bf16(
            false, A, false, Bf.v, (short)0, acc[i], false, false);
    }
#pragma unroll
    for (int i = 0; i < 4; ++i) {
        BF16Frag Bf;
        Bf.q[0] = bb[2 * i];
        Bf.q[1] = bb[2 * i + 1];
        acc[i + 4] = __builtin_amdgcn_wmma_f32_16x16x32_bf16(
            false, A, false, Bf.v, (short)0, acc[i + 4], false, false);
    }
}

#if USE_TDM
// TDM: DMA a 128-row x 32-dword f32 tile (row stride 1024 f32) from global
// into LDS, inserting 4 dwords of padding after every 32 dwords stored
// (-> LDS row stride 36 f32, matching slab[][KC+4]).
static __device__ __forceinline__ void tdm_load_tile(unsigned lds_off,
                                                     unsigned long long gaddr) {
    u32x4 g0;
    g0.x = 1u;                                        // count=1 (user descriptor)
    g0.y = lds_off;                                   // lds_addr (bytes)
    g0.z = (unsigned)gaddr;                           // global_addr[31:0]
    g0.w = (unsigned)(gaddr >> 32) | 0x80000000u;     // global_addr[56:32] | type=2<<30
    i32x8 g1;
    g1[0] = (2 << 16) | (1 << 20) | (4 << 22) | (3 << 25);  // 4B, pad 4dw per 32dw
    g1[1] = (int)(32u << 16);                         // tensor_dim0 = 32
    g1[2] = (int)(128u << 16);                        // tensor_dim1 = 128
    g1[3] = (int)(32u << 16);                         // tile_dim0 = 32
    g1[4] = 128;                                      // tile_dim1 = 128, tile_dim2 = 0
    g1[5] = 1024;                                     // tensor_dim0_stride = 1024
    g1[6] = 0;
    g1[7] = 0;
    i32x4 z4 = {0, 0, 0, 0};
#if defined(__clang_major__) && (__clang_major__ >= 23)
    i32x8 z8 = {0, 0, 0, 0, 0, 0, 0, 0};
    __builtin_amdgcn_tensor_load_to_lds(g0, g1, z4, z4, z8, 0);
#else
    __builtin_amdgcn_tensor_load_to_lds(g0, g1, z4, z4, 0);
#endif
}
#endif

// ---------------------------------------------------------------------------
// Zero accumulators (ctx_raw [B,D], den [B,H]) — must run every launch
// ---------------------------------------------------------------------------
__global__ void zero_kernel(float* ctx, float* den) {
    int i = blockIdx.x * 256 + threadIdx.x;
    if (i < B_ * D_) ctx[i] = 0.f;
    if (i < B_ * H_) den[i] = 0.f;
}

// ---------------------------------------------------------------------------
// Convert Wk / Wv (f32 [j,d] row-major) into bf16 B-fragment blocks.
// Block (kc, jt) = 32 lanes x 16 bf16, lane ln holds column n = ln&15,
// rows k = 8*(ln>>4)+e (e<8), 16+8*(ln>>4)+(e-8) (e>=8)  (ISA 16-bit B 32x16 lane layout).
// ---------------------------------------------------------------------------
__global__ void wconv_kernel(const float* __restrict__ Wk, const float* __restrict__ Wv,
                             unsigned short* __restrict__ wkb, unsigned short* __restrict__ wvb) {
    int gid  = blockIdx.x * 256 + threadIdx.x;   // 0 .. 131071
    int mat  = gid >> 16;
    int slot = gid & 0xFFFF;                     // kc*2048 + jt*32 + ln
    int kc   = slot >> 11;
    int rem  = slot & 2047;
    int jt   = rem >> 5;
    int ln   = rem & 31;
    int lane16 = ln & 15, hi = ln >> 4;

    const float* W = mat ? Wv : Wk;
    unsigned short* dst = mat ? wvb : wkb;

    int j = jt * 16 + lane16;
    const float* src = W + (size_t)j * D_ + kc * KC + 8 * hi;  // runs of 8 at 8*hi and 16+8*hi
    unsigned short* d = dst + (((size_t)(kc * JT_TOTAL + jt)) << 9) + ln * 16;
#pragma unroll
    for (int e = 0; e < 16; ++e) d[e] = f32_to_bf16(src[e < 8 ? e : e + 8]);
}

// ---------------------------------------------------------------------------
// Qp = Q @ Wq^T + bq   (tiny: 67 MFLOP, one workgroup per b)
// ---------------------------------------------------------------------------
__global__ void qp_kernel(const float* __restrict__ Q, const float* __restrict__ Wq,
                          const float* __restrict__ bq, float* __restrict__ Qp) {
    __shared__ float qrow[D_];
    const int b = blockIdx.x, t = threadIdx.x;
    for (int i = t; i < D_; i += 256) qrow[i] = Q[(size_t)b * D_ + i];
    __syncthreads();
    for (int ii = 0; ii < 4; ++ii) {
        int j = t + 256 * ii;
        const float* w = Wq + (size_t)j * D_;
        float o = bq[j];
#pragma unroll 4
        for (int d = 0; d < D_; d += 4)
            o += qrow[d] * w[d] + qrow[d + 1] * w[d + 1] +
                 qrow[d + 2] * w[d + 2] + qrow[d + 3] * w[d + 3];
        Qp[(size_t)b * D_ + j] = o;
    }
}

// ---------------------------------------------------------------------------
// Fused main kernel: per WG = (128 L-rows, 512 j-cols = 8 heads)
//   pass 1: Kp tiles via WMMA -> scores -> num = relu(s/8)^2 -> den atomics
//   pass 2: Vp tiles via WMMA -> ctx_raw += num * Vp  (global f32 atomics)
// Slabs are triple-buffered with ONE barrier per k-chunk; on TDM builds the
// next tile's DMA (TENSORcnt, in-order) overlaps the current 256 WMMAs.
// ---------------------------------------------------------------------------
__global__ void __launch_bounds__(MAIN_THREADS)
attn_main_kernel(const float* __restrict__ K, const float* __restrict__ V,
                 const unsigned short* __restrict__ wkb, const unsigned short* __restrict__ wvb,
                 const float* __restrict__ bk, const float* __restrict__ bv,
                 const float* __restrict__ Qp, float* __restrict__ ctx_raw,
                 float* __restrict__ den_g) {
    __shared__ float slab[3][RB][KC + 4];   // triple-buffered staged K/V rows (54 KB)
    __shared__ float sc[RB][8];             // scores -> num, per (l, local head)
    __shared__ float qps[JBLK];             // Qp slice for this j-block
    __shared__ float den_l[8];

    const int t      = threadIdx.x;
    const int wv32   = t >> 5;           // wave id 0..31
    const int ln     = t & 31;
    const int lane16 = ln & 15;
    const int hi     = ln >> 4;
    const int lt     = wv32 >> 2;        // l-subtile 0..7 (16 rows each)
    const int jg     = wv32 & 3;         // j-group 0..3 (8 j-tiles each)

    const int lb   = blockIdx.x;         // 0..511
    const int jb   = blockIdx.y;         // 0..1
    const size_t row0 = (size_t)lb * RB; // flattened (b*L + l)
    const int b    = (int)(row0 / L_);

    if (t < JBLK) qps[t] = Qp[(size_t)b * D_ + jb * JBLK + t];
    ((float*)sc)[t] = 0.f;
    if (t < 8) den_l[t] = 0.f;

    const int srow = t >> 3;             // fallback coop load: 128 rows x 8 float4
    const int sseg = t & 7;
    const int arow = lt * 16 + lane16;   // this lane's A-matrix row in the slab

    const v8f vzero = {0.f, 0.f, 0.f, 0.f, 0.f, 0.f, 0.f, 0.f};
    v8f acc[8];

    // per-wave B-fragment base: 8 consecutive 1KB blocks -> immediate offsets
    const size_t boff = (((size_t)(jb * JT_PER_BLK + jg * 8)) << 9) + (size_t)ln * 16;
    const size_t bstride = (size_t)JT_TOTAL << 9;   // 64 blocks * 512 ushorts per kc

#if USE_TDM
    const unsigned long long gK = (unsigned long long)(const char*)K + row0 * (D_ * 4);
    const unsigned long long gV = (unsigned long long)(const char*)V + row0 * (D_ * 4);
#endif

    // ------------------------- pass 1: Kp + scores -------------------------
#pragma unroll
    for (int i = 0; i < 8; ++i) acc[i] = vzero;

    {
        const unsigned short* bptr = wkb + boff;
#if USE_TDM
        if (wv32 == 0)
            tdm_load_tile((unsigned)(size_t)&slab[0][0][0], gK);
#else
        *(float4*)&slab[0][srow][sseg * 4] =
            *(const float4*)(K + (row0 + srow) * D_ + sseg * 4);
#endif
        int buf = 0, nbuf = 1;
        for (int kc = 0; kc < NKC; ++kc, bptr += bstride) {
#if USE_TDM
            if (wv32 == 0) {
                if (kc + 1 < NKC) {
                    tdm_load_tile((unsigned)(size_t)&slab[nbuf][0][0],
                                  gK + (unsigned long long)(kc + 1) * (KC * 4));
                    __builtin_amdgcn_s_wait_tensorcnt(1);
                } else {
                    __builtin_amdgcn_s_wait_tensorcnt(0);
                }
            }
#else
            if (kc + 1 < NKC) {
                *(float4*)&slab[nbuf][srow][sseg * 4] =
                    *(const float4*)(K + (row0 + srow) * D_ + (kc + 1) * KC + sseg * 4);
                __builtin_prefetch(K + (row0 + srow) * D_ + (kc + 2) * KC + sseg * 4, 0, 1);
            }
#endif
            __syncthreads();
            gemm_step(bptr, &slab[buf][arow][0], hi, acc);
            buf = nbuf;
            nbuf = (nbuf == 2) ? 0 : nbuf + 1;
        }
    }

    // score contributions: score[l,h] += sum_n (Kp[l,j0+n]+bk) * Qp[j0+n]
#pragma unroll
    for (int i = 0; i < 8; ++i) {
        const int jtl  = jg * 8 + i;
        const int jtg  = jb * JT_PER_BLK + jtl;
        const int jcol = jtg * 16 + lane16;
        const float bias = bk[jcol];
        const float qv   = qps[jtl * 16 + lane16];
        const int hloc   = jtl >> 2;
#pragma unroll
        for (int r = 0; r < 8; ++r) {
            float p = (acc[i][r] + bias) * qv;
            p += __shfl_xor(p, 1, 32);
            p += __shfl_xor(p, 2, 32);
            p += __shfl_xor(p, 4, 32);
            p += __shfl_xor(p, 8, 32);   // sum over the 16 n-lanes of this half
            if (lane16 == 0)
                __hip_atomic_fetch_add(&sc[lt * 16 + r + 8 * hi][hloc], p,
                                       __ATOMIC_RELAXED, __HIP_MEMORY_SCOPE_WORKGROUP);
        }
    }
    __syncthreads();

    // num = relu(score/8)^2 ; den partials
    {
        const int lrow = t >> 3, hh = t & 7;
        float s = sc[lrow][hh] * 0.125f;           // 1/sqrt(HD)
        float n = (s > 0.f) ? s * s : 0.f;         // relu^TAU, TAU=2
        sc[lrow][hh] = n;
        float p = n;
        p += __shfl_xor(p, 8, 32);
        p += __shfl_xor(p, 16, 32);                // sum over 4 l-rows per wave
        if ((ln & 24) == 0)
            __hip_atomic_fetch_add(&den_l[hh], p, __ATOMIC_RELAXED,
                                   __HIP_MEMORY_SCOPE_WORKGROUP);
    }
    __syncthreads();
    if (t < 8)
        __hip_atomic_fetch_add(&den_g[b * H_ + jb * 8 + t], den_l[t],
                               __ATOMIC_RELAXED, __HIP_MEMORY_SCOPE_AGENT);

    // ------------------------- pass 2: Vp + ctx ---------------------------
#pragma unroll
    for (int i = 0; i < 8; ++i) acc[i] = vzero;

    {
        const unsigned short* bptr = wvb + boff;
        __syncthreads();
#if USE_TDM
        if (wv32 == 0)
            tdm_load_tile((unsigned)(size_t)&slab[0][0][0], gV);
#else
        *(float4*)&slab[0][srow][sseg * 4] =
            *(const float4*)(V + (row0 + srow) * D_ + sseg * 4);
#endif
        int buf = 0, nbuf = 1;
        for (int kc = 0; kc < NKC; ++kc, bptr += bstride) {
#if USE_TDM
            if (wv32 == 0) {
                if (kc + 1 < NKC) {
                    tdm_load_tile((unsigned)(size_t)&slab[nbuf][0][0],
                                  gV + (unsigned long long)(kc + 1) * (KC * 4));
                    __builtin_amdgcn_s_wait_tensorcnt(1);
                } else {
                    __builtin_amdgcn_s_wait_tensorcnt(0);
                }
            }
#else
            if (kc + 1 < NKC) {
                *(float4*)&slab[nbuf][srow][sseg * 4] =
                    *(const float4*)(V + (row0 + srow) * D_ + (kc + 1) * KC + sseg * 4);
                __builtin_prefetch(V + (row0 + srow) * D_ + (kc + 2) * KC + sseg * 4, 0, 1);
            }
#endif
            __syncthreads();
            gemm_step(bptr, &slab[buf][arow][0], hi, acc);
            buf = nbuf;
            nbuf = (nbuf == 2) ? 0 : nbuf + 1;
        }
    }

    // ctx_raw[b,j] += sum_l num[l,h] * (Vp[l,j]+bv)   (sc holds num, read-only now)
#pragma unroll
    for (int i = 0; i < 8; ++i) {
        const int jtl  = jg * 8 + i;
        const int jtg  = jb * JT_PER_BLK + jtl;
        const int jcol = jtg * 16 + lane16;
        const float bias = bv[jcol];
        const int hloc   = jtl >> 2;
        float p = 0.f;
#pragma unroll
        for (int r = 0; r < 8; ++r)
            p += (acc[i][r] + bias) * sc[lt * 16 + r + 8 * hi][hloc];
        p += __shfl_xor(p, 16, 32);    // combine halves: rows 0..7 + 8..15
        if (ln < 16)
            __hip_atomic_fetch_add(&ctx_raw[(size_t)b * D_ + jcol], p,
                                   __ATOMIC_RELAXED, __HIP_MEMORY_SCOPE_AGENT);
    }
}

// ---------------------------------------------------------------------------
// out = LN(ctx/den^0.25 @ Wo^T + bo) * gamma + beta + Q   (one WG per b)
// ---------------------------------------------------------------------------
__global__ void out_kernel(const float* __restrict__ ctx_raw, const float* __restrict__ den_g,
                           const float* __restrict__ Wo, const float* __restrict__ bo,
                           const float* __restrict__ gamma, const float* __restrict__ beta_ln,
                           const float* __restrict__ Q, float* __restrict__ out) {
    __shared__ float ctxn[D_];
    __shared__ float obuf[D_];
    __shared__ float rd[H_];
    __shared__ float redS[8], redS2[8];
    __shared__ float mu_s, rstd_s;

    const int b = blockIdx.x, t = threadIdx.x;
    if (t < H_) rd[t] = powf(den_g[b * H_ + t] + 1e-8f, -0.25f);  // 1/(sum+eps)^BETA
    __syncthreads();
    for (int i = t; i < D_; i += 256)
        ctxn[i] = ctx_raw[(size_t)b * D_ + i] * rd[i >> 6];
    __syncthreads();

    for (int ii = 0; ii < 4; ++ii) {
        int j = t + 256 * ii;
        const float* w = Wo + (size_t)j * D_;
        float o = bo[j];
#pragma unroll 4
        for (int d = 0; d < D_; d += 4)
            o += ctxn[d] * w[d] + ctxn[d + 1] * w[d + 1] +
                 ctxn[d + 2] * w[d + 2] + ctxn[d + 3] * w[d + 3];
        obuf[j] = o;
    }
    __syncthreads();

    float s = 0.f, s2 = 0.f;
    for (int ii = 0; ii < 4; ++ii) {
        float v = obuf[t + 256 * ii];
        s += v; s2 += v * v;
    }
    for (int off = 1; off < 32; off <<= 1) {
        s  += __shfl_xor(s, off, 32);
        s2 += __shfl_xor(s2, off, 32);
    }
    if ((t & 31) == 0) { redS[t >> 5] = s; redS2[t >> 5] = s2; }
    __syncthreads();
    if (t == 0) {
        float S = 0.f, S2 = 0.f;
        for (int w = 0; w < 8; ++w) { S += redS[w]; S2 += redS2[w]; }
        float mu  = S / (float)D_;
        float var = S2 / (float)D_ - mu * mu;   // biased variance (jnp.var)
        mu_s = mu;
        rstd_s = rsqrtf(var + 1e-5f);
    }
    __syncthreads();
    for (int ii = 0; ii < 4; ++ii) {
        int j = t + 256 * ii;
        out[(size_t)b * D_ + j] =
            (obuf[j] - mu_s) * rstd_s * gamma[j] + beta_ln[j] + Q[(size_t)b * D_ + j];
    }
}

// ---------------------------------------------------------------------------
extern "C" void kernel_launch(void* const* d_in, const int* in_sizes, int n_in,
                              void* d_out, int out_size, void* d_ws, size_t ws_size,
                              hipStream_t stream) {
    const float* Q       = (const float*)d_in[0];
    const float* K       = (const float*)d_in[1];
    const float* V       = (const float*)d_in[2];
    const float* Wq      = (const float*)d_in[3];
    const float* bq      = (const float*)d_in[4];
    const float* Wk      = (const float*)d_in[5];
    const float* bk      = (const float*)d_in[6];
    const float* Wv      = (const float*)d_in[7];
    const float* bv      = (const float*)d_in[8];
    const float* Wo      = (const float*)d_in[9];
    const float* bo      = (const float*)d_in[10];
    const float* gamma   = (const float*)d_in[11];
    const float* beta_ln = (const float*)d_in[12];
    float* out = (float*)d_out;

    // workspace layout
    char* ws = (char*)d_ws;
    unsigned short* wkb = (unsigned short*)(ws);                                  // 2 MB
    unsigned short* wvb = (unsigned short*)(ws + (size_t)D_ * D_ * 2);            // 2 MB
    float* Qp      = (float*)(ws + (size_t)D_ * D_ * 4);                          // 128 KB
    float* ctx_raw = (float*)(ws + (size_t)D_ * D_ * 4 + (size_t)B_ * D_ * 4);    // 128 KB
    float* den_g   = (float*)(ws + (size_t)D_ * D_ * 4 + (size_t)2 * B_ * D_ * 4);// 2 KB

    wconv_kernel<<<512, 256, 0, stream>>>(Wk, Wv, wkb, wvb);
    zero_kernel<<<(B_ * D_ + 255) / 256, 256, 0, stream>>>(ctx_raw, den_g);
    qp_kernel<<<B_, 256, 0, stream>>>(Q, Wq, bq, Qp);
    attn_main_kernel<<<dim3((B_ * L_) / RB, NJB), MAIN_THREADS, 0, stream>>>(
        K, V, wkb, wvb, bk, bv, Qp, ctx_raw, den_g);
    out_kernel<<<B_, 256, 0, stream>>>(ctx_raw, den_g, Wo, bo, gamma, beta_ln, Q, out);
}
